// MultiPrototypeTransductiveInference_9620726743751
// MI455X (gfx1250) — matrix operand order repaired and checked by reading, not compiled, not run on hardware
//
#include <hip/hip_runtime.h>
#include <hip/hip_bf16.h>
#include <cstdint>
#include <cstddef>

// ---------------- CDNA5 WMMA types ----------------
typedef __attribute__((ext_vector_type(16))) _Float16 v16h;
typedef __attribute__((ext_vector_type(8)))  _Float16 v8h;
typedef __attribute__((ext_vector_type(8)))  float    v8f;

// ---------------- problem constants ----------------
#define FEAT_DIM   192
#define N_POINTS   2048
#define N_SUB      100
#define K_CONNECT  200
#define ALPHA_LP   0.99f
#define N_QUERIES  2
#define NUM_P      300                 // 3 * N_SUB prototypes
#define NREAL      4396                // 300 + 2*2048 graph nodes
#define NPAD       4416                // 276 * 16 (WMMA-tile padded)
#define TJ         (NPAD / 16)         // 276 tiles per dim
#define PROP_ITERS 256                 // Neumann sweeps: 0.99^256 ~ 0.077 residual
#define KSPLIT     6                   // waves cooperating on one row-tile
#define KSTEPS     (NPAD / 32 / KSPLIT) // 23 K-steps of 32 per wave (exact: 138 = 6*23)

// ---------------- WMMA fragment loaders (contiguous k = 16*(lane/16) + s, wave32) ----------------
// A-matrix 16x32 f16: lanes 0-15 -> M=lane, K = k0+{0..15}
//                     lanes 16-31 -> M=lane-16, K = k0+{16..31}
__device__ __forceinline__ v16h load_fragA(const _Float16* __restrict__ base, int ldh,
                                           int row0, int k0, int lane) {
  int m = lane & 15, hi = lane >> 4;
  const _Float16* p = base + (size_t)(row0 + m) * ldh + k0 + hi * 16;
  v8h lo = *(const v8h*)(p);
  v8h hh = *(const v8h*)(p + 8);
  v16h r;
#pragma unroll
  for (int e = 0; e < 8; ++e) { r[e] = lo[e]; r[e + 8] = hh[e]; }
  return r;
}

// B-matrix 32x16 f16: lane holds column n = lane&15; lanes 0-15 -> K=k0..k0+15,
// lanes 16-31 -> K=k0+16..k0+31.  For B[k][n] = M[col0+n][k] (row-major M), this is a
// contiguous 32B run of row (col0+n).
__device__ __forceinline__ v16h load_fragB_rows(const _Float16* __restrict__ base, int ldh,
                                                int col0, int k0, int lane) {
  int n = lane & 15, hi = lane >> 4;
  return *(const v16h*)(base + (size_t)(col0 + n) * ldh + k0 + hi * 16);
}

// ---------------- stage 1: support transpose ----------------
// support_feat [2][1][192][2048] -> Xsup [4096][192]
__global__ void k_supportT(const float* __restrict__ sf, float* __restrict__ Xsup) {
  int idx = blockIdx.x * 256 + threadIdx.x;
  if (idx >= 2 * N_POINTS * FEAT_DIM) return;
  int d = idx % FEAT_DIM;
  int r = idx / FEAT_DIM;                    // 0..4095 == c*2048+p
  int c = r >> 11, p = r & (N_POINTS - 1);
  Xsup[idx] = sf[((size_t)c * FEAT_DIM + d) * N_POINTS + p];
}

// ---------------- stage 2: masked farthest-point sampling ----------------
// group 0: background over all 4096 pts (valid = y==0)
// group 1: class0 fg over pts [0,2048)   (valid = y!=0)
// group 2: class1 fg over pts [2048,4096)
__global__ void k_fps(const float* __restrict__ Xsup, const int* __restrict__ sy,
                      int* __restrict__ seedIdx) {
  const int g = blockIdx.x;
  const int base = (g == 2) ? N_POINTS : 0;
  const int npts = (g == 0) ? 2 * N_POINTS : N_POINTS;
  const int tid = threadIdx.x;
  const float NEGF = -3.402823466e38f;
  __shared__ float mind[2 * N_POINTS];
  __shared__ float rv[1024];
  __shared__ int   ri[1024];
  __shared__ float lastf[FEAT_DIM];
  __shared__ int   s_last;

  for (int p = tid; p < npts; p += 1024) {
    int y = sy[base + p];
    bool valid = (g == 0) ? (y == 0) : (y != 0);
    mind[p] = valid ? 3.402823466e38f : NEGF;
  }
  __syncthreads();

  // start = first valid index (argmax over bool)
  int fv = npts;
  for (int p = tid; p < npts; p += 1024)
    if (mind[p] > 0.0f && p < fv) fv = p;
  ri[tid] = fv; __syncthreads();
  for (int s = 512; s > 0; s >>= 1) {
    if (tid < s && ri[tid + s] < ri[tid]) ri[tid] = ri[tid + s];
    __syncthreads();
  }
  int last = (ri[0] >= npts) ? 0 : ri[0];
  __syncthreads();

  for (int t = 0; t < N_SUB; ++t) {
    if (tid == 0) seedIdx[g * N_SUB + t] = base + last;   // global row index
    for (int d = tid; d < FEAT_DIM; d += 1024)
      lastf[d] = Xsup[(size_t)(base + last) * FEAT_DIM + d];
    __syncthreads();

    float bestv = NEGF; int besti = npts;
    for (int p = tid; p < npts; p += 1024) {
      float mv = mind[p];
      if (mv != NEGF) {                       // valid point: update min-dist
        const float* xp = Xsup + (size_t)(base + p) * FEAT_DIM;
        float d2 = 0.f;
        for (int d = 0; d < FEAT_DIM; ++d) { float df = xp[d] - lastf[d]; d2 = fmaf(df, df, d2); }
        mv = fminf(mv, d2);
        mind[p] = mv;
      }
      if (mv > bestv || (mv == bestv && p < besti)) { bestv = mv; besti = p; }
    }
    rv[tid] = bestv; ri[tid] = besti; __syncthreads();
    for (int s = 512; s > 0; s >>= 1) {
      if (tid < s) {
        if (rv[tid + s] > rv[tid] || (rv[tid + s] == rv[tid] && ri[tid + s] < ri[tid])) {
          rv[tid] = rv[tid + s]; ri[tid] = ri[tid + s];
        }
      }
      __syncthreads();
    }
    if (tid == 0) s_last = (ri[0] >= npts) ? 0 : ri[0];
    __syncthreads();
    last = s_last;
    __syncthreads();
  }
}

// ---------------- stage 3: nearest-seed assignment ----------------
__global__ void k_assign(const float* __restrict__ Xsup, const int* __restrict__ sy,
                         const int* __restrict__ seedIdx, int* __restrict__ assign) {
  int g = blockIdx.x;
  int base = (g == 2) ? N_POINTS : 0;
  int npts = (g == 0) ? 2 * N_POINTS : N_POINTS;
  for (int p = (int)threadIdx.x; p < npts; p += 256) {
    const float* xp = Xsup + (size_t)(base + p) * FEAT_DIM;
    float best = 3.402823466e38f; int bi = 0;
    for (int s = 0; s < N_SUB; ++s) {
      const float* sv = Xsup + (size_t)seedIdx[g * N_SUB + s] * FEAT_DIM;
      float d2 = 0.f;
      for (int d = 0; d < FEAT_DIM; ++d) { float df = xp[d] - sv[d]; d2 = fmaf(df, df, d2); }
      if (d2 < best) { best = d2; bi = s; }
    }
    assign[g * 4096 + p] = bi;
  }
}

// ---------------- stage 4: deterministic masked scatter-mean -> X rows 0..299 ----------------
__global__ void k_protos(const float* __restrict__ Xsup, const int* __restrict__ sy,
                         const int* __restrict__ assign, float* __restrict__ X) {
  int idx = blockIdx.x * 256 + threadIdx.x;
  if (idx >= 3 * N_SUB * FEAT_DIM) return;
  int d  = idx % FEAT_DIM;
  int gk = idx / FEAT_DIM;
  int g = gk / N_SUB, k = gk % N_SUB;
  int base = (g == 2) ? N_POINTS : 0;
  int npts = (g == 0) ? 2 * N_POINTS : N_POINTS;
  float sum = 0.f, cnt = 0.f;
  for (int p = 0; p < npts; ++p) {
    if (assign[g * 4096 + p] == k) {
      int y = sy[base + p];
      bool valid = (g == 0) ? (y == 0) : (y != 0);
      if (valid) { sum += Xsup[(size_t)(base + p) * FEAT_DIM + d]; cnt += 1.f; }
    }
  }
  int row = g * N_SUB + k;                    // bg -> 0..99, class0 -> 100.., class1 -> 200..
  X[(size_t)row * FEAT_DIM + d] = sum / fmaxf(cnt, 1.f);
}

// ---------------- stage 5: query transpose + zero padding -> X rows 300..NPAD-1 ----------------
__global__ void k_qX(const float* __restrict__ qf, float* __restrict__ X) {
  size_t idx = (size_t)blockIdx.x * 256 + threadIdx.x;
  size_t total = (size_t)(NPAD - NUM_P) * FEAT_DIM;
  if (idx >= total) return;
  int r = (int)(idx / FEAT_DIM), d = (int)(idx % FEAT_DIM);
  float v = 0.f;
  if (r < N_QUERIES * N_POINTS) {
    int q = r >> 11, p = r & (N_POINTS - 1);
    v = qf[((size_t)q * FEAT_DIM + d) * N_POINTS + p];
  }
  X[(size_t)(NUM_P + r) * FEAT_DIM + d] = v;
}

// ---------------- stage 6: f16 copy + row squared norms ----------------
__global__ void k_x2h(const float* __restrict__ X, _Float16* __restrict__ Xh,
                      float* __restrict__ x2) {
  int i = blockIdx.x, tid = threadIdx.x;
  __shared__ float s[256];
  float v = 0.f;
  if (tid < FEAT_DIM) {
    float x = X[(size_t)i * FEAT_DIM + tid];
    Xh[(size_t)i * FEAT_DIM + tid] = (_Float16)x;
    v = x * x;
  }
  s[tid] = v; __syncthreads();
  for (int st = 128; st > 0; st >>= 1) { if (tid < st) s[tid] += s[tid + st]; __syncthreads(); }
  if (tid == 0) x2[i] = s[0];
}

// ---------------- stage 7: WMMA gram -> squared L2 distance matrix ----------------
__global__ void k_gram(const _Float16* __restrict__ Xh, const float* __restrict__ x2,
                       float* __restrict__ D2) {
  int wave = (int)((blockIdx.x * blockDim.x + threadIdx.x) >> 5);
  int lane = (int)(threadIdx.x & 31);
  if (wave >= TJ * TJ) return;                 // uniform per wave -> EXEC all ones inside
  int ti = wave / TJ, tj = wave % TJ;
  int i0 = ti * 16, j0 = tj * 16;
  __builtin_prefetch(Xh + (size_t)i0 * FEAT_DIM, 0, 1);   // global_prefetch_b8
  __builtin_prefetch(Xh + (size_t)j0 * FEAT_DIM, 0, 1);
  v8f acc = {};
#pragma unroll
  for (int k0 = 0; k0 < FEAT_DIM; k0 += 32) {
    v16h a = load_fragA(Xh, FEAT_DIM, i0, k0, lane);
    v16h b = load_fragB_rows(Xh, FEAT_DIM, j0, k0, lane);
    acc = __builtin_amdgcn_wmma_f32_16x16x32_f16(false, a, false, b, (short)0, acc,
                                                 false, false);
  }
  int n = lane & 15, mhi = (lane >> 4) * 8;    // C/D layout: M = v + 8*(lane>=16), N = lane&15
  float x2j = x2[j0 + n];
#pragma unroll
  for (int v = 0; v < 8; ++v) {
    int ii = i0 + mhi + v;
    float d = x2[ii] + x2j - 2.0f * acc[v];
    D2[(size_t)ii * NPAD + (j0 + n)] = fmaxf(d, 0.0f);
  }
}

// ---------------- stage 8: per-row top-K threshold (bisection) + gaussian affinity ----------------
__global__ void k_affinity(const float* __restrict__ D2, float* __restrict__ A) {
  const int i = blockIdx.x;                    // < NREAL
  const int tid = threadIdx.x;
  const float* row = D2 + (size_t)i * NPAD;
  __shared__ float sf[256];
  __shared__ int   si[256];
  __shared__ float s_lo, s_hi;

  float mx = 0.f;
  for (int j = tid; j < NREAL; j += 256) if (j != i) mx = fmaxf(mx, row[j]);
  sf[tid] = mx; __syncthreads();
  for (int s = 128; s > 0; s >>= 1) { if (tid < s) sf[tid] = fmaxf(sf[tid], sf[tid + s]); __syncthreads(); }
  if (tid == 0) { s_lo = 0.f; s_hi = sf[0] + 1.0f; }
  __syncthreads();

  for (int it = 0; it < 30; ++it) {
    float mid = 0.5f * (s_lo + s_hi);
    int c = 0;
    for (int j = tid; j < NREAL; j += 256) if (j != i && row[j] < mid) c++;
    si[tid] = c; __syncthreads();
    for (int s = 128; s > 0; s >>= 1) { if (tid < s) si[tid] += si[tid + s]; __syncthreads(); }
    if (tid == 0) { if (si[0] > K_CONNECT) s_hi = mid; else s_lo = mid; }
    __syncthreads();
  }
  float t = s_lo;                              // count(d2 < t) <= K_CONNECT
  for (int j = tid; j < NPAD; j += 256) {
    float v = 0.f;
    if (j < NREAL && j != i) {
      float d = row[j];
      if (d < t) v = expf(-0.5f * d);          // sigma = 1
    }
    A[(size_t)i * NPAD + j] = v;
  }
}

// ---------------- stage 9: symmetrize A = A + A^T (diag already 0) ----------------
__global__ void k_sym(float* __restrict__ A) {
  size_t idx = (size_t)blockIdx.x * 256 + threadIdx.x;
  if (idx >= (size_t)NREAL * NREAL) return;
  int i = (int)(idx / NREAL), j = (int)(idx % NREAL);
  if (i < j) {
    float s = A[(size_t)i * NPAD + j] + A[(size_t)j * NPAD + i];
    A[(size_t)i * NPAD + j] = s;
    A[(size_t)j * NPAD + i] = s;
  }
}

// ---------------- stage 10: dsi = rsqrt(rowsum + eps) ----------------
__global__ void k_dsi(const float* __restrict__ A, float* __restrict__ dsi) {
  int i = blockIdx.x, tid = threadIdx.x;
  __shared__ float s[256];
  float v = 0.f;
  for (int j = tid; j < NREAL; j += 256) v += A[(size_t)i * NPAD + j];
  s[tid] = v; __syncthreads();
  for (int st = 128; st > 0; st >>= 1) { if (tid < st) s[tid] += s[tid + st]; __syncthreads(); }
  if (tid == 0) dsi[i] = rsqrtf(s[0] + 1.1920929e-7f);   // float32 eps
}

// ---------------- stage 11: Sh = f16(alpha * dsi_i * A * dsi_j), zero-padded ----------------
__global__ void k_buildS(const float* __restrict__ A, const float* __restrict__ dsi,
                         _Float16* __restrict__ Sh) {
  size_t idx = (size_t)blockIdx.x * 256 + threadIdx.x;
  if (idx >= (size_t)NPAD * NPAD) return;
  int i = (int)(idx / NPAD), j = (int)(idx % NPAD);
  float v = 0.f;
  if (i < NREAL && j < NREAL) v = ALPHA_LP * A[idx] * dsi[i] * dsi[j];
  Sh[idx] = (_Float16)v;
}

// ---------------- stage 12: label matrix Y (16-col padded), Z0 = Y ----------------
__global__ void k_Y(float* __restrict__ Y, float* __restrict__ Z) {
  int idx = blockIdx.x * 256 + threadIdx.x;
  if (idx >= NPAD * 16) return;
  int i = idx >> 4, c = idx & 15;
  float v = 0.f;
  if (i < 100 && c == 0) v = 1.f;
  else if (i >= 100 && i < 200 && c == 1) v = 1.f;
  else if (i >= 200 && i < 300 && c == 2) v = 1.f;
  Y[idx] = v; Z[idx] = v;
}

// ---------------- stage 13a: transpose-convert Z [NPAD x 16] f32 -> Zt [16 x NPAD] f16 ----------------
// Zt row n is exactly the B-fragment stream for output column n: one contiguous
// 32B v16h load per lane per K-step in k_prop (instead of 32 scalar loads + cvts).
__global__ void k_cvtZ(const float* __restrict__ Z, _Float16* __restrict__ Zt) {
  int idx = blockIdx.x * 256 + threadIdx.x;
  if (idx >= 16 * NPAD) return;
  int n = idx / NPAD, k = idx % NPAD;
  Zt[idx] = (_Float16)Z[(size_t)k * 16 + n];
}

// ---------------- stage 13b: WMMA propagation sweep  Zout = Sh * Zt^T + Y ----------------
// One block (6 waves) per 16-row strip; the 138 K-steps are split 6 x 23 across the
// waves, partial accumulators reduced through LDS. 276 blocks -> 1656 waves per sweep.
__global__ void k_prop(const _Float16* __restrict__ Sh, const _Float16* __restrict__ Zt,
                       const float* __restrict__ Y, float* __restrict__ Zout) {
  const int w    = (int)(threadIdx.x >> 5);      // 0..KSPLIT-1 (wave-uniform)
  const int lane = (int)(threadIdx.x & 31);
  const int i0   = (int)blockIdx.x * 16;
  const int kbeg = w * (KSTEPS * 32);
  const int n  = lane & 15;
  const int hi = lane >> 4;

  __builtin_prefetch(Sh + (size_t)i0 * NPAD + kbeg, 0, 1);

  v8f acc = {};
  for (int s = 0; s < KSTEPS; ++s) {
    int k0 = kbeg + s * 32;
    v16h a = load_fragA(Sh, NPAD, i0, k0, lane);
    v16h b = *(const v16h*)(Zt + (size_t)n * NPAD + k0 + hi * 16);
    acc = __builtin_amdgcn_wmma_f32_16x16x32_f16(false, a, false, b, (short)0, acc,
                                                 false, false);
  }

  __shared__ float part[KSPLIT][256];
#pragma unroll
  for (int v = 0; v < 8; ++v) part[w][v * 32 + lane] = acc[v];
  __syncthreads();

  if (w == 0) {                                   // wave-uniform branch
    int mhi = hi * 8;
#pragma unroll
    for (int v = 0; v < 8; ++v) {
      float ssum = 0.f;
#pragma unroll
      for (int u = 0; u < KSPLIT; ++u) ssum += part[u][v * 32 + lane];
      int ii = i0 + mhi + v;
      Zout[(size_t)ii * 16 + n] = ssum + Y[(size_t)ii * 16 + n];
    }
  }
}

// ---------------- stage 14: predictions + per-point CE terms ----------------
__global__ void k_pred(const float* __restrict__ Z, const int* __restrict__ qy,
                       float* __restrict__ out, float* __restrict__ lossTmp) {
  int idx = blockIdx.x * 256 + threadIdx.x;
  if (idx >= N_QUERIES * N_POINTS) return;
  int q = idx >> 11, p = idx & (N_POINTS - 1);
  const float* z = Z + (size_t)(NUM_P + idx) * 16;
  float z0 = z[0], z1 = z[1], z2 = z[2];
  out[((size_t)q * 3 + 0) * N_POINTS + p] = z0;
  out[((size_t)q * 3 + 1) * N_POINTS + p] = z1;
  out[((size_t)q * 3 + 2) * N_POINTS + p] = z2;
  float m = fmaxf(z0, fmaxf(z1, z2));
  float lse = m + logf(expf(z0 - m) + expf(z1 - m) + expf(z2 - m));
  int y = qy[idx];
  float zy = (y == 0) ? z0 : ((y == 1) ? z1 : z2);
  lossTmp[idx] = lse - zy;                     // -log p[y]
}

// deterministic final reduction (no float atomics -> replay-stable)
__global__ void k_loss(const float* __restrict__ lossTmp, float* __restrict__ out) {
  int tid = threadIdx.x;
  __shared__ float s[256];
  float v = 0.f;
  for (int i = tid; i < N_QUERIES * N_POINTS; i += 256) v += lossTmp[i];
  s[tid] = v; __syncthreads();
  for (int st = 128; st > 0; st >>= 1) { if (tid < st) s[tid] += s[tid + st]; __syncthreads(); }
  if (tid == 0) out[N_QUERIES * 3 * N_POINTS] = s[0] / (float)(N_QUERIES * N_POINTS);
}

// ---------------- host driver ----------------
extern "C" void kernel_launch(void* const* d_in, const int* in_sizes, int n_in,
                              void* d_out, int out_size, void* d_ws, size_t ws_size,
                              hipStream_t stream) {
  (void)in_sizes; (void)n_in; (void)out_size; (void)ws_size;
  const float* sf = (const float*)d_in[0];   // support_feat [2][1][192][2048]
  const int*   sy = (const int*)d_in[1];     // support_y    [2][1][2048]
  const float* qf = (const float*)d_in[2];   // query_feat   [2][192][2048]
  const int*   qy = (const int*)d_in[3];     // query_y      [2][2048]
  float* out = (float*)d_out;                // 12288 preds + 1 loss

  char* ws = (char*)d_ws;
  size_t off = 0;
  auto take = [&](size_t bytes) -> char* {
    char* p = ws + off;
    off = (off + bytes + 255) & ~(size_t)255;
    return p;
  };
  float*    Xsup    = (float*)take(sizeof(float) * (size_t)4096 * FEAT_DIM);
  float*    X       = (float*)take(sizeof(float) * (size_t)NPAD * FEAT_DIM);
  _Float16* Xh      = (_Float16*)take(sizeof(_Float16) * (size_t)NPAD * FEAT_DIM);
  float*    x2      = (float*)take(sizeof(float) * NPAD);
  int*      seedIdx = (int*)take(sizeof(int) * 3 * N_SUB);
  int*      assign  = (int*)take(sizeof(int) * 3 * 4096);
  float*    D2      = (float*)take(sizeof(float) * (size_t)NPAD * NPAD);       // 78 MB (L2-resident)
  float*    A       = (float*)take(sizeof(float) * (size_t)NPAD * NPAD);       // 78 MB
  _Float16* Sh      = (_Float16*)take(sizeof(_Float16) * (size_t)NPAD * NPAD); // 39 MB
  float*    dsi     = (float*)take(sizeof(float) * NPAD);
  float*    Y       = (float*)take(sizeof(float) * (size_t)NPAD * 16);
  float*    Za      = (float*)take(sizeof(float) * (size_t)NPAD * 16);
  float*    Zb      = (float*)take(sizeof(float) * (size_t)NPAD * 16);
  _Float16* Zt      = (_Float16*)take(sizeof(_Float16) * (size_t)16 * NPAD);
  float*    lossTmp = (float*)take(sizeof(float) * N_QUERIES * N_POINTS);

  k_supportT<<<(2 * N_POINTS * FEAT_DIM + 255) / 256, 256, 0, stream>>>(sf, Xsup);
  k_fps<<<3, 1024, 0, stream>>>(Xsup, sy, seedIdx);
  k_assign<<<3, 256, 0, stream>>>(Xsup, sy, seedIdx, assign);
  k_protos<<<(3 * N_SUB * FEAT_DIM + 255) / 256, 256, 0, stream>>>(Xsup, sy, assign, X);
  k_qX<<<(int)(((size_t)(NPAD - NUM_P) * FEAT_DIM + 255) / 256), 256, 0, stream>>>(qf, X);
  k_x2h<<<NPAD, 256, 0, stream>>>(X, Xh, x2);
  k_gram<<<(TJ * TJ * 32 + 255) / 256, 256, 0, stream>>>(Xh, x2, D2);
  k_affinity<<<NREAL, 256, 0, stream>>>(D2, A);
  k_sym<<<(int)(((size_t)NREAL * NREAL + 255) / 256), 256, 0, stream>>>(A);
  k_dsi<<<NREAL, 256, 0, stream>>>(A, dsi);
  k_buildS<<<(int)(((size_t)NPAD * NPAD + 255) / 256), 256, 0, stream>>>(A, dsi, Sh);
  k_Y<<<(NPAD * 16 + 255) / 256, 256, 0, stream>>>(Y, Za);
  for (int t = 0; t < PROP_ITERS; ++t) {
    const float* zin = (t & 1) ? Zb : Za;
    float* zout = (t & 1) ? Za : Zb;
    k_cvtZ<<<(16 * NPAD + 255) / 256, 256, 0, stream>>>(zin, Zt);
    k_prop<<<TJ, KSPLIT * 32, 0, stream>>>(Sh, Zt, Y, zout);
  }
  const float* Zf = (PROP_ITERS & 1) ? Zb : Za;   // even iter count -> Za
  k_pred<<<(N_QUERIES * N_POINTS + 255) / 256, 256, 0, stream>>>(Zf, qy, out, lossTmp);
  k_loss<<<1, 256, 0, stream>>>(lossTmp, out);
}
